// TransformerBlock_23905787969933
// MI455X (gfx1250) — matrix operand order repaired and checked by reading, not compiled, not run on hardware
//
#include <hip/hip_runtime.h>
#include <hip/hip_bf16.h>

typedef __attribute__((ext_vector_type(16))) __bf16 v16bf;
typedef __attribute__((ext_vector_type(8)))  float  v8f;
typedef unsigned short u16t;
typedef unsigned int   u32t;

#define T_TOK 4096
#define DIM   1024
#define DKV   256
#define NH    16
#define NKVH  4
#define HDIM  64
#define NEXP  8
#define FFD   2816
#define COMPD 256

__device__ __forceinline__ u16t f2bf(float f) {
  u32t u = __float_as_uint(f);
  u32t r = (u + 0x7FFFu + ((u >> 16) & 1u)) >> 16;
  return (u16t)r;
}
__device__ __forceinline__ float bf2f(u16t h) {
  return __uint_as_float(((u32t)h) << 16);
}

// ---------------------------------------------------------------------------
// Tiled bf16 WMMA GEMM: C[M,N] = A[M,K] @ B[K,N] (+bias) (+f32 accumulate)
// Block tile 128x128, 8 waves (4x2), each wave 32x64 = 2x4 WMMA 16x16x32 tiles.
// Double-buffered LDS pipeline; A fragment reads are explicit b128 loads; B is gathered per element.
// ---------------------------------------------------------------------------
#define BMT 128
#define BNT 128
#define BKT 32
#define LDA 40   // 32 + 8 pad (bf16 elements, 80B row stride: 16B aligned)
#define LDB 136  // 128 + 8 pad (272B row stride: 16B aligned)

template <bool HAS_BIAS, bool HAS_ACC, bool OUT_F, bool OUT_H>
__global__ __launch_bounds__(256) void gemm_bf16_kernel(
    const u16t* __restrict__ A, const u16t* __restrict__ B,
    float* __restrict__ outF, u16t* __restrict__ outH,
    const float* __restrict__ bias, const float* __restrict__ accum,
    int M, int N, int K)
{
  __shared__ u16t lsA[2][BMT * LDA];
  __shared__ u16t lsB[2][BKT * LDB];
  const int tid   = threadIdx.x;
  const int lane  = tid & 31;
  const int wave  = tid >> 5;
  const int wm    = wave >> 1;           // 0..3
  const int wn    = wave & 1;            // 0..1
  const int bm    = blockIdx.x * BMT;
  const int bn    = blockIdx.y * BNT;
  const int rowHalf = (lane >> 4) << 3;  // 0 or 8
  const int nlane   = lane & 15;

  v8f c[2][4];
  for (int mt = 0; mt < 2; ++mt)
    for (int nt = 0; nt < 4; ++nt)
      for (int r = 0; r < 8; ++r) {
        if (HAS_ACC) {
          int row = bm + wm * 32 + mt * 16 + r + rowHalf;
          int col = bn + wn * 64 + nt * 16 + nlane;
          c[mt][nt][r] = accum[(size_t)row * N + col];
        } else {
          c[mt][nt][r] = 0.f;
        }
      }

  const int arow = tid >> 1;          // 0..127
  const int acol = (tid & 1) * 16;    // 0 / 16
  const int brow = tid >> 3;          // 0..31
  const int bcol = (tid & 7) * 16;    // 0..112

  const u16t* apr = A + (size_t)(bm + arow) * K + acol;
  const u16t* bpr = B + (size_t)brow * N + bn + bcol;

  // ---- prologue: load K-tile 0 into LDS buffer 0
  {
    uint4 a0 = ((const uint4*)apr)[0];
    uint4 a1 = ((const uint4*)apr)[1];
    uint4 b0 = ((const uint4*)bpr)[0];
    uint4 b1 = ((const uint4*)bpr)[1];
    ((uint4*)&lsA[0][arow * LDA + acol])[0]     = a0;
    ((uint4*)&lsA[0][arow * LDA + acol + 8])[0] = a1;
    ((uint4*)&lsB[0][brow * LDB + bcol])[0]     = b0;
    ((uint4*)&lsB[0][brow * LDB + bcol + 8])[0] = b1;
  }
  __syncthreads();

  int buf = 0;
  for (int k0 = 0; k0 < K; k0 += BKT) {
    const bool has_next = (k0 + BKT) < K;
    uint4 na0, na1, nb0, nb1;
    if (has_next) {  // next tile's global loads in flight during this tile's WMMAs
      const u16t* ap = apr + k0 + BKT;
      const u16t* bp = bpr + (size_t)(k0 + BKT) * N;
      na0 = ((const uint4*)ap)[0];
      na1 = ((const uint4*)ap)[1];
      nb0 = ((const uint4*)bp)[0];
      nb1 = ((const uint4*)bp)[1];
      if (k0 + 2 * BKT < K) {  // gfx1250 global_prefetch_b8
        __builtin_prefetch(ap + BKT, 0, 3);
        __builtin_prefetch(bp + (size_t)BKT * N, 0, 3);
      }
    }

    v16bf af[2];
    v16bf bfrag[4];
    for (int mt = 0; mt < 2; ++mt) {
      uint4* u = (uint4*)&af[mt];
      int row = wm * 32 + mt * 16 + nlane;
      u[0] = *(const uint4*)&lsA[buf][row * LDA + rowHalf];
      u[1] = *(const uint4*)&lsA[buf][row * LDA + 16 + rowHalf];
    }
    for (int nt = 0; nt < 4; ++nt) {
      u16t* p = (u16t*)&bfrag[nt];
      int colb = wn * 64 + nt * 16 + nlane;
      for (int s = 0; s < 16; ++s) {
        int kk = (s < 8) ? (rowHalf + s) : (8 + rowHalf + s);
        p[s] = lsB[buf][kk * LDB + colb];
      }
    }
#pragma unroll
    for (int mt = 0; mt < 2; ++mt)
#pragma unroll
      for (int nt = 0; nt < 4; ++nt)
        c[mt][nt] = __builtin_amdgcn_wmma_f32_16x16x32_bf16(
            false, af[mt], false, bfrag[nt], (short)0, c[mt][nt], false, false);

    if (has_next) {
      int nb = buf ^ 1;
      ((uint4*)&lsA[nb][arow * LDA + acol])[0]     = na0;
      ((uint4*)&lsA[nb][arow * LDA + acol + 8])[0] = na1;
      ((uint4*)&lsB[nb][brow * LDB + bcol])[0]     = nb0;
      ((uint4*)&lsB[nb][brow * LDB + bcol + 8])[0] = nb1;
      __syncthreads();
      buf = nb;
    }
  }

  for (int mt = 0; mt < 2; ++mt)
    for (int nt = 0; nt < 4; ++nt) {
      int col = bn + wn * 64 + nt * 16 + nlane;
      float badd = HAS_BIAS ? bias[col] : 0.f;
      for (int r = 0; r < 8; ++r) {
        int row = bm + wm * 32 + mt * 16 + r + rowHalf;
        float v = c[mt][nt][r] + badd;
        size_t idx = (size_t)row * N + col;
        if (OUT_F) outF[idx] = v;
        if (OUT_H) outH[idx] = f2bf(v);
      }
    }
}

// ---------------------------------------------------------------------------
// K transpose: kT[b][kvh][feat(64)][key(1024)] <- k_h[b*1024+key][kvh*64+feat]
// ---------------------------------------------------------------------------
__global__ __launch_bounds__(256) void ktranspose_kernel(
    const u16t* __restrict__ kh, u16t* __restrict__ kT)
{
  int i = blockIdx.x * 256 + threadIdx.x;   // 4*1024*256 = 1,048,576
  int feat = i & (DKV - 1);
  int tok  = i >> 8;
  int b = tok >> 10, key = tok & 1023;
  int kvh = feat >> 6, f = feat & (HDIM - 1);
  kT[(((size_t)(b * NKVH + kvh) * HDIM + f) << 10) + key] = kh[(size_t)tok * DKV + feat];
}

// ---------------------------------------------------------------------------
// Flash attention: one wave per (b, h, 16-query tile). bf16 WMMA for QK^T & PV.
// 64-key superblocks: one 16-lane max butterfly per row per 64 keys; the
// softmax row-sum comes from an all-ones-B WMMA (no shuffles, lane-uniform).
// ---------------------------------------------------------------------------
#define PLD 72  // P staging row stride (elements) = 144B: 16B aligned

__global__ __launch_bounds__(32) void attn_kernel(
    const u16t* __restrict__ qh, const u16t* __restrict__ kT,
    const u16t* __restrict__ vh, u16t* __restrict__ oh)
{
  __shared__ u16t pst[16 * PLD];
  const int lane = threadIdx.x;
  const int qt = blockIdx.x;   // 0..63
  const int h  = blockIdx.y;   // 0..15
  const int b  = blockIdx.z;   // 0..3
  const int kvh = h >> 2;
  const int rowHalf = (lane >> 4) << 3;
  const int nlane   = lane & 15;
  const int qrow0 = b * 1024 + qt * 16;
  const float scale = 0.125f;  // 1/sqrt(64)

  // Q tile [16 x 64] as two 16x32 A fragments (2 x b128 each)
  v16bf aq[2];
  for (int cch = 0; cch < 2; ++cch) {
    uint4* u = (uint4*)&aq[cch];
    const u16t* qrow = qh + (size_t)(qrow0 + nlane) * DIM + h * HDIM + cch * 32;
    u[0] = *(const uint4*)&qrow[rowHalf];
    u[1] = *(const uint4*)&qrow[16 + rowHalf];
  }

  // all-ones B fragment (bf16 1.0): row-sum generator for the softmax l stat
  v16bf bones;
  {
    u32t* u = (u32t*)&bones;
#pragma unroll
    for (int j = 0; j < 8; ++j) u[j] = 0x3F803F80u;
  }

  v8f o[4], lacc;
  for (int nt = 0; nt < 4; ++nt)
    for (int r = 0; r < 8; ++r) o[nt][r] = 0.f;
  for (int r = 0; r < 8; ++r) lacc[r] = 0.f;
  float mR[8];
  for (int r = 0; r < 8; ++r) mR[r] = -3.0e38f;

  const u16t* kt0 = kT + ((size_t)(b * NKVH + kvh) * HDIM) * 1024;
  const u16t* kt1 = kT + ((size_t)(b * NKVH + kvh) * HDIM + 32) * 1024;

  for (int kb = 0; kb < 1024; kb += 64) {   // 64-key superblock
    // ---- scores: 4 tiles of 16 keys
    v8f s[4];
#pragma unroll
    for (int blk = 0; blk < 4; ++blk) {
      int kstart = kb + blk * 16;
      v16bf bk0, bk1;
      {
        u16t* p0 = (u16t*)&bk0;
        u16t* p1 = (u16t*)&bk1;
        for (int s = 0; s < 16; ++s) {
          int kk = (s < 8) ? (rowHalf + s) : (8 + rowHalf + s);
          p0[s] = kt0[(size_t)kk * 1024 + kstart + nlane];
          p1[s] = kt1[(size_t)kk * 1024 + kstart + nlane];
        }
      }
      v8f acc;
      for (int r = 0; r < 8; ++r) acc[r] = 0.f;
      acc = __builtin_amdgcn_wmma_f32_16x16x32_bf16(
          false, aq[0], false, bk0, (short)0, acc, false, false);
      acc = __builtin_amdgcn_wmma_f32_16x16x32_bf16(
          false, aq[1], false, bk1, (short)0, acc, false, false);
      s[blk] = acc;
    }

    // ---- online softmax over 64 keys: one butterfly per row
#pragma unroll
    for (int r = 0; r < 8; ++r) {
      float x0 = s[0][r] * scale;
      float x1 = s[1][r] * scale;
      float x2 = s[2][r] * scale;
      float x3 = s[3][r] * scale;
      float mx = fmaxf(fmaxf(x0, x1), fmaxf(x2, x3));
      mx = fmaxf(mx, __shfl_xor(mx, 1, 16));
      mx = fmaxf(mx, __shfl_xor(mx, 2, 16));
      mx = fmaxf(mx, __shfl_xor(mx, 4, 16));
      mx = fmaxf(mx, __shfl_xor(mx, 8, 16));
      float mnew = fmaxf(mR[r], mx);
      float alpha = __expf(mR[r] - mnew);
      mR[r] = mnew;
      float p0 = __expf(x0 - mnew);
      float p1 = __expf(x1 - mnew);
      float p2 = __expf(x2 - mnew);
      float p3 = __expf(x3 - mnew);
      lacc[r] *= alpha;
#pragma unroll
      for (int nt = 0; nt < 4; ++nt) o[nt][r] *= alpha;
      int prow = r + rowHalf;
      pst[prow * PLD + nlane]      = f2bf(p0);
      pst[prow * PLD + 16 + nlane] = f2bf(p1);
      pst[prow * PLD + 32 + nlane] = f2bf(p2);
      pst[prow * PLD + 48 + nlane] = f2bf(p3);
    }
    __syncthreads();

    // ---- P fragments (2 chunks of 32 keys), l via ones-WMMA, then PV
#pragma unroll
    for (int ch = 0; ch < 2; ++ch) {
      v16bf pa;
      {
        uint4* u = (uint4*)&pa;
        u[0] = *(const uint4*)&pst[nlane * PLD + ch * 32 + rowHalf];
        u[1] = *(const uint4*)&pst[nlane * PLD + ch * 32 + 16 + rowHalf];
      }
      lacc = __builtin_amdgcn_wmma_f32_16x16x32_bf16(
          false, pa, false, bones, (short)0, lacc, false, false);
      const u16t* vrow =
          vh + (size_t)(b * 1024 + kb + ch * 32) * DKV + kvh * HDIM;
#pragma unroll
      for (int nt = 0; nt < 4; ++nt) {
        v16bf bv;
        u16t* p = (u16t*)&bv;
        for (int s = 0; s < 16; ++s) {
          int kk = (s < 8) ? (rowHalf + s) : (8 + rowHalf + s);
          p[s] = vrow[(size_t)kk * DKV + nt * 16 + nlane];
        }
        o[nt] = __builtin_amdgcn_wmma_f32_16x16x32_bf16(
            false, pa, false, bv, (short)0, o[nt], false, false);
      }
    }
    __syncthreads();
  }

#pragma unroll
  for (int r = 0; r < 8; ++r) {
    float inv = 1.f / lacc[r];   // ones-WMMA made this lane-uniform per row
    int row = qrow0 + r + rowHalf;
#pragma unroll
    for (int nt = 0; nt < 4; ++nt) {
      int col = h * HDIM + nt * 16 + nlane;
      oh[(size_t)row * DIM + col] = f2bf(o[nt][r] * inv);
    }
  }
}

// ---------------------------------------------------------------------------
// Small kernels
// ---------------------------------------------------------------------------
__global__ __launch_bounds__(256) void rmsnorm_kernel(
    const float* __restrict__ x, const float* __restrict__ w,
    u16t* __restrict__ outH)
{
  __shared__ float red[256];
  const int t = blockIdx.x;
  const int tid = threadIdx.x;
  const float* xr = x + (size_t)t * DIM;
  float s = 0.f;
  for (int i = tid; i < DIM; i += 256) { float v = xr[i]; s += v * v; }
  red[tid] = s;
  __syncthreads();
  for (int o = 128; o > 0; o >>= 1) {
    if (tid < o) red[tid] += red[tid + o];
    __syncthreads();
  }
  float rinv = rsqrtf(red[0] / (float)DIM + 1e-6f);
  for (int i = tid; i < DIM; i += 256)
    outH[(size_t)t * DIM + i] = f2bf(w[i] * xr[i] * rinv);
}

__global__ __launch_bounds__(256) void f2bf_kernel(
    const float* __restrict__ in, u16t* __restrict__ out, int n)
{
  int i = blockIdx.x * 256 + threadIdx.x;
  if (i < n) out[i] = f2bf(in[i]);
}

__global__ __launch_bounds__(256) void add_kernel(
    const float* __restrict__ a, const float* __restrict__ b,
    float* __restrict__ outF, u16t* __restrict__ outH, int n)
{
  int i = blockIdx.x * 256 + threadIdx.x;
  if (i < n) {
    float v = a[i] + b[i];
    outF[i] = v;
    if (outH) outH[i] = f2bf(v);
  }
}

__global__ __launch_bounds__(128) void gate_kernel(
    const u16t* __restrict__ h2n, const float* __restrict__ gw,
    float* __restrict__ comb)
{
  int t = blockIdx.x * 128 + threadIdx.x;
  float acc[NEXP];
#pragma unroll
  for (int e = 0; e < NEXP; ++e) acc[e] = 0.f;
  const u16t* xr = h2n + (size_t)t * DIM;
  for (int d = 0; d < DIM; ++d) {
    float a = bf2f(xr[d]);
    const float* g = gw + d * NEXP;
#pragma unroll
    for (int e = 0; e < NEXP; ++e) acc[e] += a * g[e];
  }
  float m = acc[0];
#pragma unroll
  for (int e = 1; e < NEXP; ++e) m = fmaxf(m, acc[e]);
  float p[NEXP], se = 0.f;
#pragma unroll
  for (int e = 0; e < NEXP; ++e) { p[e] = __expf(acc[e] - m); se += p[e]; }
#pragma unroll
  for (int e = 0; e < NEXP; ++e) p[e] /= se;
  int i1 = 0;
#pragma unroll
  for (int e = 1; e < NEXP; ++e) if (p[e] > p[i1]) i1 = e;
  int i2 = (i1 == 0) ? 1 : 0;
#pragma unroll
  for (int e = 0; e < NEXP; ++e) if (e != i1 && p[e] > p[i2]) i2 = e;
  float s = p[i1] + p[i2];
#pragma unroll
  for (int e = 0; e < NEXP; ++e)
    comb[(size_t)t * NEXP + e] = (e == i1) ? p[i1] / s : ((e == i2) ? p[i2] / s : 0.f);
}

__global__ __launch_bounds__(256) void act_kernel(
    const u16t* __restrict__ h1, const u16t* __restrict__ h3,
    const float* __restrict__ comb, int e, u16t* __restrict__ act)
{
  size_t i = (size_t)blockIdx.x * 256 + threadIdx.x;
  if (i >= (size_t)T_TOK * FFD) return;
  int t = (int)(i / FFD);
  float a = bf2f(h1[i]);
  float bb = bf2f(h3[i]);
  float s = a / (1.f + __expf(-a));
  act[i] = f2bf(comb[(size_t)t * NEXP + e] * s * bb);
}

// ---------------------------------------------------------------------------
extern "C" void kernel_launch(void* const* d_in, const int* in_sizes, int n_in,
                              void* d_out, int out_size, void* d_ws, size_t ws_size,
                              hipStream_t stream)
{
  (void)in_sizes; (void)n_in; (void)out_size; (void)ws_size;
  const float* hidden = (const float*)d_in[0];
  /* d_in[1] = position_ids (unused) */
  const float* prev  = (const float*)d_in[2];
  const float* wq    = (const float*)d_in[3];
  const float* wk    = (const float*)d_in[4];
  const float* wv    = (const float*)d_in[5];
  const float* wo    = (const float*)d_in[6];
  const float* mixw  = (const float*)d_in[7];
  const float* mixb  = (const float*)d_in[8];
  const float* gatew = (const float*)d_in[9];
  const float* w1    = (const float*)d_in[10];
  const float* w2    = (const float*)d_in[11];
  const float* w3    = (const float*)d_in[12];
  const float* ln1   = (const float*)d_in[13];
  const float* ln2   = (const float*)d_in[14];
  const float* compw = (const float*)d_in[15];
  const float* compb = (const float*)d_in[16];

  float* out_h     = (float*)d_out;                     // [T,D]
  float* out_mixed = out_h + (size_t)T_TOK * DIM;       // [T,D]
  float* out_comp  = out_mixed + (size_t)T_TOK * DIM;   // [T,COMP]

  size_t off = 0;
  auto alloc = [&](size_t bytes) -> void* {
    void* p = (char*)d_ws + off;
    off += (bytes + 255) & ~(size_t)255;
    return p;
  };
  auto aU = [&](size_t n) -> u16t*  { return (u16t*)alloc(n * sizeof(u16t)); };
  auto aF = [&](size_t n) -> float* { return (float*)alloc(n * sizeof(float)); };

  // bf16 weight copies
  u16t* wq_h    = aU((size_t)DIM * DIM);
  u16t* wk_h    = aU((size_t)DIM * DKV);
  u16t* wv_h    = aU((size_t)DIM * DKV);
  u16t* wo_h    = aU((size_t)DIM * DIM);
  u16t* mixw_h  = aU((size_t)2 * DIM * DIM);
  u16t* w1_h    = aU((size_t)NEXP * DIM * FFD);
  u16t* w2_h    = aU((size_t)NEXP * FFD * DIM);
  u16t* w3_h    = aU((size_t)NEXP * DIM * FFD);
  u16t* compw_h = aU((size_t)DIM * COMPD);
  // bf16 activations
  u16t* xn_h    = aU((size_t)T_TOK * DIM);
  u16t* q_h     = aU((size_t)T_TOK * DIM);
  u16t* k_h     = aU((size_t)T_TOK * DKV);
  u16t* v_h     = aU((size_t)T_TOK * DKV);
  u16t* kT_h    = aU((size_t)T_TOK * DKV);   // transposed K
  u16t* attn_h  = aU((size_t)T_TOK * DIM);
  u16t* cur_h   = aU((size_t)T_TOK * DIM);
  u16t* prev_h  = aU((size_t)T_TOK * DIM);
  u16t* h2n_h   = aU((size_t)T_TOK * DIM);
  u16t* hfin_h  = aU((size_t)T_TOK * DIM);
  u16t* h1_h    = aU((size_t)T_TOK * FFD);
  u16t* h3_h    = aU((size_t)T_TOK * FFD);
  u16t* act_h   = aU((size_t)T_TOK * FFD);
  // f32 scratch
  float* h_f   = aF((size_t)T_TOK * DIM);
  float* moe_f = aF((size_t)T_TOK * DIM);
  float* comb  = aF((size_t)T_TOK * NEXP);

  auto conv = [&](const float* src, u16t* dst, size_t n) {
    f2bf_kernel<<<dim3((unsigned)((n + 255) / 256)), 256, 0, stream>>>(src, dst, (int)n);
  };
  auto gemm = [&](const u16t* A, const u16t* B, float* oF, u16t* oH,
                  const float* bias, const float* acc, int M, int N, int K) {
    dim3 g((unsigned)(M / BMT), (unsigned)(N / BNT));
    if (oH && !oF)
      gemm_bf16_kernel<false, false, false, true><<<g, 256, 0, stream>>>(
          A, B, oF, oH, bias, acc, M, N, K);
    else if (bias)
      gemm_bf16_kernel<true, false, true, false><<<g, 256, 0, stream>>>(
          A, B, oF, oH, bias, acc, M, N, K);
    else if (acc)
      gemm_bf16_kernel<false, true, true, false><<<g, 256, 0, stream>>>(
          A, B, oF, oH, bias, acc, M, N, K);
    else
      gemm_bf16_kernel<false, false, true, false><<<g, 256, 0, stream>>>(
          A, B, oF, oH, bias, acc, M, N, K);
  };

  // 1) weight + prev conversions
  conv(wq, wq_h, (size_t)DIM * DIM);
  conv(wk, wk_h, (size_t)DIM * DKV);
  conv(wv, wv_h, (size_t)DIM * DKV);
  conv(wo, wo_h, (size_t)DIM * DIM);
  conv(mixw, mixw_h, (size_t)2 * DIM * DIM);
  conv(w1, w1_h, (size_t)NEXP * DIM * FFD);
  conv(w2, w2_h, (size_t)NEXP * FFD * DIM);
  conv(w3, w3_h, (size_t)NEXP * DIM * FFD);
  conv(compw, compw_h, (size_t)DIM * COMPD);
  conv(prev, prev_h, (size_t)T_TOK * DIM);

  // 2) x = rmsnorm(hidden, ln1)
  rmsnorm_kernel<<<T_TOK, 256, 0, stream>>>(hidden, ln1, xn_h);

  // 3) q/k/v projections (+ K transpose for WMMA-friendly layout)
  gemm(xn_h, wq_h, nullptr, q_h, nullptr, nullptr, T_TOK, DIM, DIM);
  gemm(xn_h, wk_h, nullptr, k_h, nullptr, nullptr, T_TOK, DKV, DIM);
  gemm(xn_h, wv_h, nullptr, v_h, nullptr, nullptr, T_TOK, DKV, DIM);
  ktranspose_kernel<<<(T_TOK * DKV) / 256, 256, 0, stream>>>(k_h, kT_h);

  // 4) attention
  attn_kernel<<<dim3(64, 16, 4), 32, 0, stream>>>(q_h, kT_h, v_h, attn_h);

  // 5) output projection
  gemm(attn_h, wo_h, nullptr, cur_h, nullptr, nullptr, T_TOK, DIM, DIM);

  // 6/7) mixed = cur @ mixW[:D] + prev @ mixW[D:] + mix_b
  gemm(cur_h, mixw_h, out_mixed, nullptr, mixb, nullptr, T_TOK, DIM, DIM);
  gemm(prev_h, mixw_h + (size_t)DIM * DIM, out_mixed, nullptr, nullptr, out_mixed,
       T_TOK, DIM, DIM);

  // 8) h = hidden + mixed ; 9) h2n = rmsnorm(h, ln2)
  add_kernel<<<(T_TOK * DIM) / 256, 256, 0, stream>>>(hidden, out_mixed, h_f, nullptr,
                                                      T_TOK * DIM);
  rmsnorm_kernel<<<T_TOK, 256, 0, stream>>>(h_f, ln2, h2n_h);

  // 10) gating (softmax + top-2 renormalized combine weights)
  gate_kernel<<<T_TOK / 128, 128, 0, stream>>>(h2n_h, gatew, comb);

  // 11) MoE: per-expert GLU, combine folded into activation, accumulate w2 GEMMs
  for (int e = 0; e < NEXP; ++e) {
    gemm(h2n_h, w1_h + (size_t)e * DIM * FFD, nullptr, h1_h, nullptr, nullptr,
         T_TOK, FFD, DIM);
    gemm(h2n_h, w3_h + (size_t)e * DIM * FFD, nullptr, h3_h, nullptr, nullptr,
         T_TOK, FFD, DIM);
    act_kernel<<<(T_TOK * FFD) / 256, 256, 0, stream>>>(h1_h, h3_h, comb, e, act_h);
    gemm(act_h, w2_h + (size_t)e * FFD * DIM, moe_f, nullptr, nullptr,
         (e == 0) ? nullptr : moe_f, T_TOK, DIM, FFD);
  }

  // 12) h_final = h + moe
  add_kernel<<<(T_TOK * DIM) / 256, 256, 0, stream>>>(h_f, moe_f, out_h, hfin_h,
                                                      T_TOK * DIM);

  // 13) layer_compressed = h_final @ comp_w + comp_b
  gemm(hfin_h, compw_h, out_comp, nullptr, compb, nullptr, T_TOK, COMPD, DIM);
}
